// StackedLSTM_58935541235731
// MI455X (gfx1250) — compile-verified, matrix operand order checked by reading, not run on hardware
//
#include <hip/hip_runtime.h>

// ---------------------------------------------------------------------------
// StackedLSTM (T=256, B=32, EMB=512, HID=1024, VOCAB=32000) for gfx1250.
// bf16 WMMA (v_wmma_f32_16x16x32_bf16) with f32 accumulation.
// GEMMs use 64x64 register tiles (16 WMMAs / k-step, 2.0 wmma per fragment
// load); __launch_bounds__(256,1) + streamed B-fragments keep the live set
// under the 256-VGPR budget (no scratch spills).
// ---------------------------------------------------------------------------

#define T_STEPS 256
#define BATCH   32
#define EMB     512
#define HID     1024
#define VOCAB   32000
#define ROWS    (T_STEPS * BATCH)   // 8192
#define G4H     (4 * HID)           // 4096

typedef __bf16  bf16_t;
typedef bf16_t  bf16x16 __attribute__((ext_vector_type(16)));
typedef float   f32x8   __attribute__((ext_vector_type(8)));

// ---- scalar helpers --------------------------------------------------------

__device__ __forceinline__ unsigned short f32_to_bf16_u16(float f) {
    unsigned int u = __float_as_uint(f);
    u += 0x7FFFu + ((u >> 16) & 1u);     // round to nearest even
    return (unsigned short)(u >> 16);
}

__device__ __forceinline__ float sigmoidf_fast(float x) {
    return 1.0f / (1.0f + __expf(-x));
}

// Load a 16-element A/B fragment (16x32 bf16, row-major source, this lane's
// row already folded into `p`; p points at [row][kb], 4-byte aligned).
// Lane<16: elems 0..7 = K kb+0..7, elems 8..15 = K 16+kb..16+kb+7.
__device__ __forceinline__ bf16x16 load_frag16(const unsigned short* __restrict__ p) {
    const unsigned int* q = (const unsigned int*)p;
    bf16x16 f;
#pragma unroll
    for (int j = 0; j < 4; ++j) {
        unsigned int w = q[j];
        f[2 * j]     = __builtin_bit_cast(bf16_t, (unsigned short)(w & 0xFFFFu));
        f[2 * j + 1] = __builtin_bit_cast(bf16_t, (unsigned short)(w >> 16));
    }
#pragma unroll
    for (int j = 0; j < 4; ++j) {
        unsigned int w = q[8 + j];
        f[8 + 2 * j] = __builtin_bit_cast(bf16_t, (unsigned short)(w & 0xFFFFu));
        f[9 + 2 * j] = __builtin_bit_cast(bf16_t, (unsigned short)(w >> 16));
    }
    return f;
}

// ---- small utility kernels -------------------------------------------------

__global__ void zero_u32_kernel(unsigned int* __restrict__ p, int n) {
    int i = blockIdx.x * 256 + threadIdx.x;
    if (i < n) p[i] = 0u;
}

// x0[t*B+b][e] = bf16(embedding[data[t*B+b]][e])
__global__ void embed_kernel(const int* __restrict__ data,
                             const float* __restrict__ emb,
                             unsigned short* __restrict__ x0) {
    int i = blockIdx.x * 256 + threadIdx.x;           // < ROWS*EMB
    if (i >= ROWS * EMB) return;
    int row = i >> 9;                                  // /512
    int e   = i & (EMB - 1);
    int tok = data[row];
    x0[i] = f32_to_bf16_u16(emb[(long long)tok * EMB + e]);
}

// Wt[n][k] = bf16(W[k][n]);  W: [K,N] f32 row-major -> Wt: [N,K] bf16 row-major
__global__ void transpose_bf16_kernel(const float* __restrict__ W,
                                      unsigned short* __restrict__ Wt,
                                      int K, int N) {
    long long i = (long long)blockIdx.x * 256 + threadIdx.x;
    long long total = (long long)K * N;
    if (i >= total) return;
    int n = (int)(i / K);
    int k = (int)(i % K);
    Wt[i] = f32_to_bf16_u16(W[(long long)k * N + n]);
}

// ---- generic bf16 WMMA GEMM: C[M,N] = A[M,K]*Bt[N,K]^T + bias --------------
// One wave computes a 64x64 tile. Per 32-deep k-step: 4 A-frags stay live,
// B-frags are streamed one at a time, each feeding 4 WMMAs (16 WMMAs total).
// M, N must be multiples of 64; K a multiple of 32.

__global__ void __launch_bounds__(256, 1)
gemm_bf16_kernel(const unsigned short* __restrict__ A, int lda,
                 const unsigned short* __restrict__ Bt, int ldb,
                 const float* __restrict__ bias,
                 float* __restrict__ C, int ldc,
                 int M, int N, int K) {
    const int lane   = threadIdx.x & 31;
    const int wave   = threadIdx.x >> 5;
    const long long gw = (long long)blockIdx.x * 8 + wave;
    const int nTiles = N >> 6;
    const int mTiles = M >> 6;
    if (gw >= (long long)nTiles * mTiles) return;
    const int mT = (int)(gw / nTiles);   // waves in a block share mT -> A hits WGP$
    const int nT = (int)(gw % nTiles);
    const int m0 = mT << 6;
    const int n0 = nT << 6;

    const int r  = lane & 15;
    const int kb = (lane & 16) ? 8 : 0;

    const unsigned short* arow = A  + (long long)(m0 + r) * lda + kb;
    const unsigned short* brow = Bt + (long long)(n0 + r) * ldb + kb;
    const long long astep = (long long)16 * lda;
    const long long bstep = (long long)16 * ldb;

    f32x8 acc[4][4] = {};
    for (int k = 0; k < K; k += 32) {
        bf16x16 af0 = load_frag16(arow + 0 * astep + k);
        bf16x16 af1 = load_frag16(arow + 1 * astep + k);
        bf16x16 af2 = load_frag16(arow + 2 * astep + k);
        bf16x16 af3 = load_frag16(arow + 3 * astep + k);
#pragma unroll
        for (int ni = 0; ni < 4; ++ni) {
            bf16x16 bf = load_frag16(brow + ni * bstep + k);
            acc[0][ni] = __builtin_amdgcn_wmma_f32_16x16x32_bf16(
                false, af0, false, bf, (short)0, acc[0][ni], false, false);
            acc[1][ni] = __builtin_amdgcn_wmma_f32_16x16x32_bf16(
                false, af1, false, bf, (short)0, acc[1][ni], false, false);
            acc[2][ni] = __builtin_amdgcn_wmma_f32_16x16x32_bf16(
                false, af2, false, bf, (short)0, acc[2][ni], false, false);
            acc[3][ni] = __builtin_amdgcn_wmma_f32_16x16x32_bf16(
                false, af3, false, bf, (short)0, acc[3][ni], false, false);
        }
    }

    const int ccol = lane & 15;
    const int rofs = (lane >> 4) << 3;
#pragma unroll
    for (int mi = 0; mi < 4; ++mi) {
#pragma unroll
        for (int ni = 0; ni < 4; ++ni) {
            const int cn = n0 + ni * 16 + ccol;
            const float bv = bias ? bias[cn] : 0.0f;
            const long long rbase = (long long)(m0 + mi * 16 + rofs) * ldc + cn;
#pragma unroll
            for (int e = 0; e < 8; ++e)
                C[rbase + (long long)e * ldc] = acc[mi][ni][e] + bv;
        }
    }
}

// ---- recurrent step: gates = Gx[t] + h_in @ Wh ; LSTM cell update ----------
// Grid: 16 blocks (64 hidden units each) x 256 threads (8 waves).
// Each wave: 4 tiles of 16x16 gate outputs (2 batch row-tiles x 16 col-tiles
// spanning the i/g/f/o blocks for this 64-unit slice).

__global__ void lstm_step_kernel(const float* __restrict__ Gx,          // [ROWS, 4096]
                                 const unsigned short* __restrict__ WhT, // rows [4096], ld=ldWh (k offset pre-applied)
                                 int ldWh,
                                 const unsigned short* __restrict__ h_in,  // [32,1024] bf16
                                 unsigned short* __restrict__ h_out,       // [32,1024] bf16
                                 float* __restrict__ c_state,              // [32,1024] f32
                                 unsigned short* __restrict__ out,         // [ROWS,1024] bf16
                                 int t) {
    __shared__ float lds[BATCH * 256];       // 32 batch x (4 gates x 64 cols) = 32 KB

    const int lane = threadIdx.x & 31;
    const int wave = threadIdx.x >> 5;
    const int hc0  = blockIdx.x * 64;
    const int r    = lane & 15;
    const int kb   = (lane & 16) ? 8 : 0;

#pragma unroll
    for (int j = 0; j < 4; ++j) {
        const int tau  = wave * 4 + j;
        const int m0   = (tau & 1) << 4;     // batch row tile: 0 or 16
        const int ct   = tau >> 1;           // 0..15
        const int gate = ct >> 2;            // i,g,f,o
        const int sub  = ct & 3;
        const int n    = gate * HID + hc0 + sub * 16;

        // seed accumulator with Gx (already includes input-GEMM + bias)
        f32x8 acc;
        const int crow0 = t * BATCH + m0 + ((lane >> 4) << 3);
        const int ccol  = n + (lane & 15);
#pragma unroll
        for (int e = 0; e < 8; ++e)
            acc[e] = Gx[(long long)(crow0 + e) * G4H + ccol];

        const unsigned short* arow = h_in + (m0 + r) * HID + kb;
        const unsigned short* brow = WhT + (long long)(n + r) * ldWh + kb;
        for (int k = 0; k < HID; k += 32) {
            bf16x16 af = load_frag16(arow + k);
            bf16x16 bf = load_frag16(brow + k);
            acc = __builtin_amdgcn_wmma_f32_16x16x32_bf16(false, af, false, bf, (short)0, acc, false, false);
        }

        const int lrow0 = m0 + ((lane >> 4) << 3);
        const int lcol  = gate * 64 + sub * 16 + (lane & 15);
#pragma unroll
        for (int e = 0; e < 8; ++e)
            lds[(lrow0 + e) * 256 + lcol] = acc[e];
    }

    __syncthreads();

    // cell update: 32 batch x 64 hidden = 2048 elems, 8 per thread
    for (int u = threadIdx.x; u < BATCH * 64; u += 256) {
        const int b  = u >> 6;
        const int hl = u & 63;
        const float gi = lds[b * 256 + hl];
        const float gg = lds[b * 256 + 64 + hl];
        const float gf = lds[b * 256 + 128 + hl];
        const float go = lds[b * 256 + 192 + hl];
        const int hg = hc0 + hl;
        const float cp = c_state[b * HID + hg];
        const float cn = sigmoidf_fast(gf + 1.0f) * cp + sigmoidf_fast(gi) * tanhf(gg);
        const float hn = sigmoidf_fast(go) * tanhf(cn);
        c_state[b * HID + hg] = cn;
        const unsigned short hb = f32_to_bf16_u16(hn);
        h_out[b * HID + hg] = hb;
        out[((long long)(t * BATCH + b)) * HID + hg] = hb;
    }
}

// ---------------------------------------------------------------------------

extern "C" void kernel_launch(void* const* d_in, const int* in_sizes, int n_in,
                              void* d_out, int out_size, void* d_ws, size_t ws_size,
                              hipStream_t stream) {
    (void)in_sizes; (void)n_in; (void)out_size; (void)ws_size;

    const int*   data = (const int*)d_in[0];
    const float* emb  = (const float*)d_in[2];
    const float* Wl[3] = {(const float*)d_in[3], (const float*)d_in[5], (const float*)d_in[7]};
    const float* bl[3] = {(const float*)d_in[4], (const float*)d_in[6], (const float*)d_in[8]};
    const float* Wd   = (const float*)d_in[9];
    const float* bd   = (const float*)d_in[10];
    float* logits = (float*)d_out;

    const int Din[3]  = {EMB, HID, HID};
    const int Ktot[3] = {EMB + HID, HID + HID, HID + HID};

    // ---- workspace layout (256B aligned), ~305 MB total ----
    char* ws = (char*)d_ws;
    size_t off = 0;
    auto take = [&](size_t bytes) -> char* {
        char* p = ws + off;
        off = (off + bytes + 255) & ~(size_t)255;
        return p;
    };
    unsigned short* x0   = (unsigned short*)take((size_t)ROWS * EMB * 2);
    unsigned short* outb[3];
    for (int l = 0; l < 3; ++l) outb[l] = (unsigned short*)take((size_t)ROWS * HID * 2);
    unsigned short* WT[3];
    for (int l = 0; l < 3; ++l) WT[l] = (unsigned short*)take((size_t)G4H * Ktot[l] * 2);
    unsigned short* WdT = (unsigned short*)take((size_t)VOCAB * HID * 2);
    float* Gx = (float*)take((size_t)ROWS * G4H * 4);
    // c_state (f32) followed immediately by ping-pong h_state (bf16)
    char* state = take((size_t)BATCH * HID * 4 + 2 * (size_t)BATCH * HID * 2);
    float*          c_state = (float*)state;
    unsigned short* h_state = (unsigned short*)(state + (size_t)BATCH * HID * 4);

    // ---- 1) embedding gather -> bf16 ----
    embed_kernel<<<(ROWS * EMB + 255) / 256, 256, 0, stream>>>(data, emb, x0);

    // ---- 2) weight transpose + bf16 convert ----
    for (int l = 0; l < 3; ++l) {
        long long tot = (long long)Ktot[l] * G4H;
        transpose_bf16_kernel<<<(unsigned)((tot + 255) / 256), 256, 0, stream>>>(
            Wl[l], WT[l], Ktot[l], G4H);
    }
    {
        long long tot = (long long)HID * VOCAB;
        transpose_bf16_kernel<<<(unsigned)((tot + 255) / 256), 256, 0, stream>>>(
            Wd, WdT, HID, VOCAB);
    }

    // ---- 3) layers ----
    for (int l = 0; l < 3; ++l) {
        // zero c_state + both h_state buffers
        int nz = (BATCH * HID * 4 + 2 * BATCH * HID * 2) / 4;
        zero_u32_kernel<<<(nz + 255) / 256, 256, 0, stream>>>((unsigned int*)state, nz);

        // Gx = Xl @ Wx + b  (A: [8192, Din], Bt: Wx^T rows of WT[l], K=Din)
        const unsigned short* Ain = (l == 0) ? x0 : outb[l - 1];
        const int ldaA = Din[l];
        {
            long long tiles = (long long)(ROWS / 64) * (G4H / 64);
            gemm_bf16_kernel<<<(unsigned)((tiles + 7) / 8), 256, 0, stream>>>(
                Ain, ldaA, WT[l], Ktot[l], bl[l], Gx, G4H, ROWS, G4H, Din[l]);
        }

        // sequential scan; WhT rows start at k-offset Din[l]
        const unsigned short* WhT = WT[l] + Din[l];
        for (int t = 0; t < T_STEPS; ++t) {
            const unsigned short* hin  = h_state + (size_t)(t & 1) * BATCH * HID;
            unsigned short*       hout = h_state + (size_t)((t + 1) & 1) * BATCH * HID;
            lstm_step_kernel<<<16, 256, 0, stream>>>(
                Gx, WhT, Ktot[l], hin, hout, c_state, outb[l], t);
        }
    }

    // ---- 4) decoder: logits = out2 @ Wd + bd ----
    {
        long long tiles = (long long)(ROWS / 64) * (VOCAB / 64);
        gemm_bf16_kernel<<<(unsigned)((tiles + 7) / 8), 256, 0, stream>>>(
            outb[2], HID, WdT, HID, bd, logits, VOCAB, ROWS, VOCAB, HID);
    }
}
